// VisionRetention_37357625540639
// MI455X (gfx1250) — matrix operand order repaired and checked by reading, not compiled, not run on hardware
//
#include <hip/hip_runtime.h>
#include <cstddef>
#include <cstdint>

typedef __bf16 bf16;
typedef __attribute__((ext_vector_type(16))) __bf16 v16bf;
typedef __attribute__((ext_vector_type(8)))  float  v8f;

#define WMMA_BF16(a, b, c) \
    __builtin_amdgcn_wmma_f32_16x16x32_bf16(false, (a), false, (b), (short)0, (c), false, false)

static __device__ __forceinline__ v8f zero_v8f() {
    v8f z;
#pragma unroll
    for (int i = 0; i < 8; i++) z[i] = 0.0f;
    return z;
}
static __device__ __forceinline__ v16bf zero_v16bf() {
    v16bf z;
#pragma unroll
    for (int i = 0; i < 16; i++) z[i] = (bf16)0.0f;
    return z;
}

// ---------------------------------------------------------------------------
// Elementwise helpers
// ---------------------------------------------------------------------------
__global__ void f32_to_bf16_kernel(const float* __restrict__ in, bf16* __restrict__ out, long n) {
    long i = (long)blockIdx.x * blockDim.x + threadIdx.x;
    if (i < n) out[i] = (bf16)in[i];
}

// x: [32,3,224,224] -> xp bf16 [32*196, 768] with (c, py, px) flattening
__global__ void im2col_kernel(const float* __restrict__ x, bf16* __restrict__ xp) {
    long i = (long)blockIdx.x * blockDim.x + threadIdx.x;
    const long total = (long)32 * 196 * 768;
    if (i >= total) return;
    int e = (int)(i % 768);
    long m = i / 768;
    int b = (int)(m / 196), p = (int)(m % 196);
    int gy = p / 14, gx = p % 14;
    int c = e >> 8;           // /256
    int r = (e >> 4) & 15;    // row within patch
    int cc = e & 15;          // col within patch
    long src = (((long)b * 3 + c) * 224 + (gy * 16 + r)) * 224 + (gx * 16 + cc);
    xp[i] = (bf16)x[src];
}

__global__ void cls_init_kernel(const float* __restrict__ cls, float* __restrict__ tok) {
    int i = blockIdx.x * blockDim.x + threadIdx.x;
    if (i >= 32 * 768) return;
    int b = i / 768, c = i % 768;
    tok[((long)b * 197 + 196) * 768 + c] = cls[c];
}

// One block (256 threads) per row of 768; writes bf16 normalized row.
__global__ __launch_bounds__(256) void layernorm_kernel(
    const float* __restrict__ x, const float* __restrict__ w, const float* __restrict__ bia,
    bf16* __restrict__ out, int row_base, int row_stride) {
    int r = blockIdx.x;
    long in_row = (long)row_base + (long)r * row_stride;
    const float* xr = x + in_row * 768;
    int tid = threadIdx.x;
    float v0 = xr[tid], v1 = xr[tid + 256], v2 = xr[tid + 512];
    float s = v0 + v1 + v2;
    float ss = v0 * v0 + v1 * v1 + v2 * v2;
#pragma unroll
    for (int off = 16; off >= 1; off >>= 1) {
        s += __shfl_xor(s, off, 32);
        ss += __shfl_xor(ss, off, 32);
    }
    __shared__ float sred[16];
    int wv = tid >> 5;
    if ((tid & 31) == 0) { sred[wv] = s; sred[8 + wv] = ss; }
    __syncthreads();
    float ts = 0.0f, tss = 0.0f;
#pragma unroll
    for (int i = 0; i < 8; i++) { ts += sred[i]; tss += sred[8 + i]; }
    float mu = ts * (1.0f / 768.0f);
    float var = tss * (1.0f / 768.0f) - mu * mu;
    float rs = rsqrtf(var + 1e-6f);
    long orow = (long)r * 768;
    out[orow + tid]       = (bf16)((v0 - mu) * rs * w[tid]       + bia[tid]);
    out[orow + tid + 256] = (bf16)((v1 - mu) * rs * w[tid + 256] + bia[tid + 256]);
    out[orow + tid + 512] = (bf16)((v2 - mu) * rs * w[tid + 512] + bia[tid + 512]);
}

// ---------------------------------------------------------------------------
// Generic WMMA GEMM: D[M,Nout] = A[M,K](bf16,row-major) * W[Nout,K]^T + epilogue
// One wave -> 16x64 tile (4 accumulators). Block = 8 waves = 128 rows.
// All M values used here are multiples of 16, so the row guard is wave-uniform;
// the column guard is wave-uniform too -> fast unguarded path for full tiles.
// mode: 0 = bf16 out (+bias)           1 = bf16 out gelu(+bias)
//       2 = f32 resid += (acc+bias)*ls 3 = f32 out (+bias)
//       4 = f32 out (+bias+pos), row remap 196->197 (patch embed)
// ---------------------------------------------------------------------------
__global__ __launch_bounds__(256) void gemm_bf16_kernel(
    const bf16* __restrict__ A, int lda,
    const bf16* __restrict__ W, int K,
    const float* __restrict__ bias,
    bf16* __restrict__ out_bf, float* __restrict__ out_f32,
    const float* __restrict__ ls, const float* __restrict__ pos,
    int M, int Nout, int ldo, int mode) {
    int wave = threadIdx.x >> 5;
    int lane = threadIdx.x & 31;
    int half = lane >> 4;
    int l16  = lane & 15;
    int m0 = (blockIdx.y * 8 + wave) * 16;
    if (m0 >= M) return;
    int n0 = blockIdx.x * 64;

    v8f acc[4];
#pragma unroll
    for (int t = 0; t < 4; t++) acc[t] = zero_v8f();

    int am = m0 + l16;
    const bf16* Arow = A + (size_t)am * lda + half * 16;
    bool fullTile = (n0 + 64 <= Nout) && (m0 + 16 <= M);  // wave-uniform

    if (fullTile) {
        const bf16* Wr[4];
#pragma unroll
        for (int t = 0; t < 4; t++)
            Wr[t] = W + (size_t)(n0 + t * 16 + l16) * K + half * 16;
        for (int k0 = 0; k0 < K; k0 += 32) {
            v16bf a;
            const bf16* p = Arow + k0;
#pragma unroll
            for (int i = 0; i < 16; i++) a[i] = p[i];
            __builtin_prefetch(p + 64, 0, 1);  // speculative: global_prefetch_b8
#pragma unroll
            for (int t = 0; t < 4; t++) {
                const bf16* q = Wr[t] + k0;
                v16bf bw;
#pragma unroll
                for (int i = 0; i < 16; i++) bw[i] = q[i];
                acc[t] = WMMA_BF16(a, bw, acc[t]);
            }
        }
    } else {
        for (int k0 = 0; k0 < K; k0 += 32) {
            v16bf a = zero_v16bf();
            if (am < M) {
                const bf16* p = Arow + k0;
#pragma unroll
                for (int i = 0; i < 16; i++) a[i] = p[i];
            }
#pragma unroll
            for (int t = 0; t < 4; t++) {
                int n = n0 + t * 16 + l16;
                v16bf bw = zero_v16bf();
                if (n < Nout) {
                    const bf16* q = W + (size_t)n * K + k0 + half * 16;
#pragma unroll
                    for (int i = 0; i < 16; i++) bw[i] = q[i];
                }
                acc[t] = WMMA_BF16(a, bw, acc[t]);
            }
        }
    }

    // Epilogue. C layout: lane l16 = col, vgpr r -> row m0 + r + half*8
#pragma unroll
    for (int t = 0; t < 4; t++) {
        int col = n0 + t * 16 + l16;
        if (col >= Nout) continue;
        float bv = bias ? bias[col] : 0.0f;
#pragma unroll
        for (int r = 0; r < 8; r++) {
            int row = m0 + r + half * 8;
            if (row >= M) continue;
            float v = acc[t][r] + bv;
            if (mode == 0) {
                out_bf[(size_t)row * ldo + col] = (bf16)v;
            } else if (mode == 1) {
                float g = 0.5f * v * (1.0f + erff(v * 0.70710678118f));
                out_bf[(size_t)row * ldo + col] = (bf16)g;
            } else if (mode == 2) {
                out_f32[(size_t)row * ldo + col] += v * ls[col];
            } else if (mode == 3) {
                out_f32[(size_t)row * ldo + col] = v;
            } else {
                int orow = (row / 196) * 197 + (row % 196);
                out_f32[(size_t)orow * ldo + col] = v + pos[(size_t)(row % 196) * 768 + col];
            }
        }
    }
}

// ---------------------------------------------------------------------------
// Fused retention attention. One block (4 waves) per (batch, head).
// qkv: bf16 [32*197, 2304]  (q | k | v, head-major within each 768 chunk)
// out: bf16 [32*197, 768]
// V is staged into LDS with GLOBAL_LOAD_ASYNC_TO_LDS_B128 (ASYNCcnt path).
// ---------------------------------------------------------------------------
__global__ __launch_bounds__(128) void attention_kernel(
    const bf16* __restrict__ qkv, bf16* __restrict__ out) {
    __shared__ bf16 v_lds[224][64];       // staged V for this (b,h): 128B rows
    __shared__ bf16 p_lds[4][16][224];    // per-wave probability tile
    int b = blockIdx.x / 12;
    int h = blockIdx.x % 12;
    int wave = threadIdx.x >> 5;
    int lane = threadIdx.x & 31;
    int half = lane >> 4, l16 = lane & 15;

    const bf16* qb = qkv + (size_t)b * 197 * 2304 + (size_t)h * 64;
    const bf16* kb = qb + 768;
    const bf16* vb = qb + 1536;

    // Async copy V rows (197 x 64 bf16 = 197 x 8 x 16B chunks) into LDS.
    {
        unsigned lbase = (unsigned)(uintptr_t)(&v_lds[0][0]);  // LDS byte offset
        for (int e = threadIdx.x; e < 197 * 8; e += 128) {
            int key = e >> 3, seg = e & 7;
            const bf16* g = vb + (size_t)key * 2304 + seg * 8;
            unsigned loff = lbase + (unsigned)(key * 128 + seg * 16);
            asm volatile("global_load_async_to_lds_b128 %0, %1, off"
                         :: "v"(loff), "v"(g)
                         : "memory");
        }
        // zero-fill pad rows 197..223 with direct LDS stores
        for (int e = threadIdx.x; e < 27 * 64; e += 128) {
            int key = 197 + (e >> 6), d = e & 63;
            v_lds[key][d] = (bf16)0.0f;
        }
        asm volatile("s_wait_asynccnt 0x0" ::: "memory");
    }
    __syncthreads();

    const float loga = -3.46573590f;   // ln(1/32)
    const float logb = -6.23832463f;   // ln(1/512)
    float gamma = 1.0f - __expf(loga + (logb - loga) * ((float)h * (1.0f / 11.0f)));
    float lg = __logf(gamma);

    for (int qt = wave; qt < 13; qt += 4) {
        int q0 = qt * 16;
        int qm = q0 + l16;
        // Q fragments for the two 32-wide d halves
        v16bf qa[2];
#pragma unroll
        for (int dh = 0; dh < 2; dh++) {
            v16bf a = zero_v16bf();
            if (qm < 197) {
                const bf16* p = qb + (size_t)qm * 2304 + dh * 32 + half * 16;
#pragma unroll
                for (int i = 0; i < 16; i++) a[i] = p[i];
            }
            qa[dh] = a;
        }
        // scores for all 13 key tiles, kept in registers
        v8f sc[13];
#pragma unroll
        for (int kt = 0; kt < 13; kt++) {
            v8f acc = zero_v8f();
            int key = kt * 16 + l16;
#pragma unroll
            for (int dh = 0; dh < 2; dh++) {
                v16bf bw = zero_v16bf();
                if (key < 197) {
                    const bf16* p = kb + (size_t)key * 2304 + dh * 32 + half * 16;
#pragma unroll
                    for (int i = 0; i < 16; i++) bw[i] = p[i];
                }
                acc = WMMA_BF16(qa[dh], bw, acc);
            }
            // decay mask: * SCALE * gamma^(i-j) for j<=i; exact 0 above diag; -inf pad
#pragma unroll
            for (int r = 0; r < 8; r++) {
                int i = q0 + r + half * 8;
                int j = kt * 16 + l16;
                float s;
                if (j >= 197)   s = -1e30f;
                else if (j > i) s = 0.0f;
                else            s = acc[r] * (0.125f * __expf(lg * (float)(i - j)));
                sc[kt][r] = s;
            }
        }
        // softmax along each row (16 lanes of the half hold one tile row)
#pragma unroll
        for (int r = 0; r < 8; r++) {
            float mx = -1e30f;
#pragma unroll
            for (int kt = 0; kt < 13; kt++) mx = fmaxf(mx, sc[kt][r]);
#pragma unroll
            for (int off = 8; off >= 1; off >>= 1) mx = fmaxf(mx, __shfl_xor(mx, off, 32));
            float sum = 0.0f;
#pragma unroll
            for (int kt = 0; kt < 13; kt++) {
                float e = __expf(sc[kt][r] - mx);
                sc[kt][r] = e;
                sum += e;
            }
#pragma unroll
            for (int off = 8; off >= 1; off >>= 1) sum += __shfl_xor(sum, off, 32);
            float inv = 1.0f / sum;
#pragma unroll
            for (int kt = 0; kt < 13; kt++) sc[kt][r] *= inv;
        }
        // P -> LDS (pad cols 208..223 with zeros)
#pragma unroll
        for (int z = 0; z < 8; z++) {
            int e = lane * 8 + z;
            p_lds[wave][e >> 4][208 + (e & 15)] = (bf16)0.0f;
        }
#pragma unroll
        for (int kt = 0; kt < 13; kt++) {
#pragma unroll
            for (int r = 0; r < 8; r++) {
                p_lds[wave][r + half * 8][kt * 16 + l16] = (bf16)sc[kt][r];
            }
        }
        // out(16x64) = P(16x224) @ V(224x64)
        v8f oacc[4];
#pragma unroll
        for (int t = 0; t < 4; t++) oacc[t] = zero_v8f();
        for (int ks = 0; ks < 7; ks++) {
            v16bf pa;
#pragma unroll
            for (int i = 0; i < 16; i++) {
                pa[i] = p_lds[wave][l16][ks * 32 + half * 16 + i];
            }
#pragma unroll
            for (int dt = 0; dt < 4; dt++) {
                v16bf bw;
#pragma unroll
                for (int i = 0; i < 16; i++) {
                    bw[i] = v_lds[ks * 32 + half * 16 + i][dt * 16 + l16];
                }
                oacc[dt] = WMMA_BF16(pa, bw, oacc[dt]);
            }
        }
        // store out tile
#pragma unroll
        for (int dt = 0; dt < 4; dt++) {
#pragma unroll
            for (int r = 0; r < 8; r++) {
                int i = q0 + r + half * 8;
                if (i < 197) {
                    out[((size_t)b * 197 + i) * 768 + h * 64 + dt * 16 + l16] = (bf16)oacc[dt][r];
                }
            }
        }
    }
}

// ---------------------------------------------------------------------------
// Host orchestration
// ---------------------------------------------------------------------------
extern "C" void kernel_launch(void* const* d_in, const int* in_sizes, int n_in,
                              void* d_out, int out_size, void* d_ws, size_t ws_size,
                              hipStream_t stream) {
    (void)in_sizes; (void)n_in; (void)out_size; (void)ws_size;
    const float* x       = (const float*)d_in[0];
    const float* patch_w = (const float*)d_in[1];
    const float* patch_b = (const float*)d_in[2];
    const float* pos     = (const float*)d_in[3];
    const float* cls     = (const float*)d_in[4];
    const float* n1w     = (const float*)d_in[5];
    const float* n1b     = (const float*)d_in[6];
    const float* qkv_w   = (const float*)d_in[7];
    const float* qkv_b   = (const float*)d_in[8];
    const float* proj_w  = (const float*)d_in[9];
    const float* proj_b  = (const float*)d_in[10];
    const float* ls1     = (const float*)d_in[11];
    const float* n2w     = (const float*)d_in[12];
    const float* n2b     = (const float*)d_in[13];
    const float* fc1_w   = (const float*)d_in[14];
    const float* fc1_b   = (const float*)d_in[15];
    const float* fc2_w   = (const float*)d_in[16];
    const float* fc2_b   = (const float*)d_in[17];
    const float* ls2     = (const float*)d_in[18];
    const float* nfw     = (const float*)d_in[19];
    const float* nfb     = (const float*)d_in[20];
    const float* head_w  = (const float*)d_in[21];
    const float* head_b  = (const float*)d_in[22];

    const int M = 32 * 197;  // 6304 token rows
    char* ws = (char*)d_ws;
    size_t off = 0;
    auto alloc = [&](size_t bytes) -> void* {
        void* p = ws + off;
        off += (bytes + 255) & ~(size_t)255;
        return p;
    };
    float* tok = (float*)alloc((size_t)M * 768 * 4);        // fp32 residual stream
    bf16*  hbf = (bf16*) alloc((size_t)M * 768 * 2);        // LN output / pooled
    bf16*  atn = (bf16*) alloc((size_t)M * 768 * 2);        // attention output
    bf16*  big = (bf16*) alloc((size_t)M * 3072 * 2);       // xp / qkv / fc1-out
    bf16*  wbf = (bf16*) alloc((size_t)3072 * 768 * 2);     // per-GEMM bf16 weights

    auto cvt = [&](const float* src, bf16* dst, long n) {
        f32_to_bf16_kernel<<<(unsigned)((n + 255) / 256), 256, 0, stream>>>(src, dst, n);
    };
    auto gemm = [&](const bf16* A, int lda, const bf16* W, int K, const float* bias,
                    bf16* obf, float* of32, const float* lsp, const float* posp,
                    int Mm, int Nout, int ldo, int mode) {
        dim3 g((Nout + 63) / 64, (Mm + 127) / 128);
        gemm_bf16_kernel<<<g, 256, 0, stream>>>(A, lda, W, K, bias, obf, of32, lsp, posp,
                                                Mm, Nout, ldo, mode);
    };

    // Patch embed: im2col -> GEMM(+bias+pos, row remap) -> cls row
    im2col_kernel<<<(32 * 196 * 768 + 255) / 256, 256, 0, stream>>>(x, big);
    cvt(patch_w, wbf, (long)768 * 768);
    gemm(big, 768, wbf, 768, patch_b, nullptr, tok, nullptr, pos, 32 * 196, 768, 768, 4);
    cls_init_kernel<<<(32 * 768 + 255) / 256, 256, 0, stream>>>(cls, tok);

    for (int l = 0; l < 12; l++) {
        layernorm_kernel<<<M, 256, 0, stream>>>(tok, n1w + l * 768, n1b + l * 768, hbf, 0, 1);
        cvt(qkv_w + (size_t)l * 2304 * 768, wbf, (long)2304 * 768);
        gemm(hbf, 768, wbf, 768, qkv_b + l * 2304, big, nullptr, nullptr, nullptr,
             M, 2304, 2304, 0);
        attention_kernel<<<32 * 12, 128, 0, stream>>>(big, atn);
        cvt(proj_w + (size_t)l * 768 * 768, wbf, (long)768 * 768);
        gemm(atn, 768, wbf, 768, proj_b + l * 768, nullptr, tok, ls1 + l * 768, nullptr,
             M, 768, 768, 2);
        layernorm_kernel<<<M, 256, 0, stream>>>(tok, n2w + l * 768, n2b + l * 768, hbf, 0, 1);
        cvt(fc1_w + (size_t)l * 3072 * 768, wbf, (long)3072 * 768);
        gemm(hbf, 768, wbf, 768, fc1_b + l * 3072, big, nullptr, nullptr, nullptr,
             M, 3072, 3072, 1);
        cvt(fc2_w + (size_t)l * 768 * 3072, wbf, (long)768 * 3072);
        gemm(big, 3072, wbf, 3072, fc2_b + l * 768, nullptr, tok, ls2 + l * 768, nullptr,
             M, 768, 768, 2);
    }

    // Final LN on the 32 cls rows -> head GEMM -> d_out fp32 [32,1000]
    layernorm_kernel<<<32, 256, 0, stream>>>(tok, nfw, nfb, hbf, 196, 197);
    cvt(head_w, wbf, (long)1000 * 768);
    gemm(hbf, 768, wbf, 768, head_b, nullptr, (float*)d_out, nullptr, nullptr,
         32, 1000, 1000, 3);
}
